// GBST_79121887526925
// MI455X (gfx1250) — matrix operand order repaired and checked by reading, not compiled, not run on hardware
//
#include <hip/hip_runtime.h>
#include <hip/hip_bf16.h>

#define BATCH 4
#define SEQ   4096
#define DIM   512
#define KW    4
#define LPAD  4104            // ceil(4096/12)*12, divisible by 1,2,3,4
#define MROWS (BATCH*SEQ)     // 16384

typedef __attribute__((ext_vector_type(16))) __bf16 v16bf;
typedef __attribute__((ext_vector_type(8)))  __bf16 v8bf;
typedef __attribute__((ext_vector_type(4)))  __bf16 v4bf;
typedef __attribute__((ext_vector_type(8)))  float  v8f;

// ---------------------------------------------------------------------------
// 1) Fused embedding gather + depthwise conv (K=4, zero-padded tail) -> bf16 Y
//    Y layout: (B*SEQ, DIM) row-major, d contiguous.
// ---------------------------------------------------------------------------
__global__ __launch_bounds__(128) void conv_embed_kernel(
    const int* __restrict__ x, const float* __restrict__ emb,
    const float* __restrict__ dww, const float* __restrict__ dwb,
    __bf16* __restrict__ Y)
{
    const int g  = blockIdx.x;            // 0..MROWS-1
    const int b  = g >> 12;               // /SEQ
    const int t  = g & (SEQ - 1);
    const int d0 = threadIdx.x * 4;       // 128 threads * 4 = 512

    float w[4][4];
#pragma unroll
    for (int i = 0; i < 4; ++i) {
        float4 wv = *(const float4*)(dww + (size_t)(d0 + i) * KW);
        w[i][0] = wv.x; w[i][1] = wv.y; w[i][2] = wv.z; w[i][3] = wv.w;
    }
    float acc[4] = {0.f, 0.f, 0.f, 0.f};
#pragma unroll
    for (int k = 0; k < KW; ++k) {
        const int tt = t + k;
        if (tt < SEQ) {
            const int tok = x[b * SEQ + tt];
            float4 e = *(const float4*)(emb + (size_t)tok * DIM + d0);
            acc[0] += e.x * w[0][k];
            acc[1] += e.y * w[1][k];
            acc[2] += e.z * w[2][k];
            acc[3] += e.w * w[3][k];
        }
    }
    v4bf o;
#pragma unroll
    for (int i = 0; i < 4; ++i) o[i] = (__bf16)(acc[i] + dwb[d0 + i]);
    *(v4bf*)(Y + (size_t)g * DIM + d0) = o;
}

// ---------------------------------------------------------------------------
// 2) pw_w f32 -> bf16 (no transpose needed: B-fragment wants fixed-N rows
//    with contiguous K, which is exactly pw_w[e][d] row-major).
// ---------------------------------------------------------------------------
__global__ __launch_bounds__(256) void cvt_pw_kernel(
    const float* __restrict__ pw, __bf16* __restrict__ Wb)
{
    const int i = blockIdx.x * 256 + threadIdx.x;
    if (i < DIM * DIM) Wb[i] = (__bf16)pw[i];
}

// ---------------------------------------------------------------------------
// 3) Zero the pad rows (t = SEQ..LPAD-1) of the f32 projected buffer YP(B,L,D)
// ---------------------------------------------------------------------------
__global__ __launch_bounds__(256) void zeropad_kernel(float* __restrict__ YP)
{
    const int idx = blockIdx.x * 256 + threadIdx.x;     // 4*8*512 = 16384
    if (idx >= BATCH * (LPAD - SEQ) * DIM) return;
    const int b = idx / ((LPAD - SEQ) * DIM);
    const int r = (idx / DIM) % (LPAD - SEQ);
    const int d = idx % DIM;
    YP[((size_t)b * LPAD + SEQ + r) * DIM + d] = 0.f;
}

// ---------------------------------------------------------------------------
// 4) WMMA GEMM: YP[b,t,e] = sum_d Y[b,t,d] * pw_w[e,d] + pw_b[e]
//    Wave32 bf16 WMMA, f32 accumulate.
//    - A tile (16 rows x 512 d = 16 KB) staged once per block into LDS via
//      GLOBAL_LOAD_ASYNC_TO_LDS_B128 (ASYNCcnt), then read with ds_load_b128.
//    - Each wave computes 16x64 output (4 accumulators reusing one A frag).
//    - B (pw_w bf16) loaded direct from global: 0.5 MB, L2-resident, reused
//      by all blocks.
// ---------------------------------------------------------------------------
__global__ __launch_bounds__(256) void gemm_pw_kernel(
    const __bf16* __restrict__ Y, const __bf16* __restrict__ Wb,
    const float* __restrict__ pwb, float* __restrict__ YP)
{
    __shared__ __bf16 tileA[16 * DIM];    // 16 KB

    const int lane  = threadIdx.x & 31;
    const int wave  = threadIdx.x >> 5;   // 0..7
    const int mtile = blockIdx.x;         // 0..1023
    const int nbase = wave * 64;          // 8 waves x 64 cols = 512
    const int half  = lane >> 4;
    const int mlane = lane & 15;

    // ---- async stage of the A tile: 16 KB = 1024 x b128, 4 chunks/thread ----
    {
        const char* src  = (const char*)(Y + (size_t)mtile * 16 * DIM);
        char*       dstb = (char*)tileA;
#pragma unroll
        for (int i = 0; i < 4; ++i) {
            const int byt = (threadIdx.x + i * 256) * 16;
            // generic LDS pointer low 32 bits == LDS byte offset (aperture rule)
            unsigned int lds_addr = (unsigned int)(uintptr_t)(dstb + byt);
            unsigned long long gaddr = (unsigned long long)(uintptr_t)(src + byt);
            asm volatile("global_load_async_to_lds_b128 %0, %1, off"
                         :: "v"(lds_addr), "v"(gaddr) : "memory");
        }
        asm volatile("s_wait_asynccnt 0x0" ::: "memory");
    }
    __syncthreads();

    const __bf16* arow = tileA + (size_t)mlane * DIM;
    const __bf16* bp[4];
#pragma unroll
    for (int j = 0; j < 4; ++j)
        bp[j] = Wb + (size_t)(nbase + j * 16 + mlane) * DIM;

    // warm the B panel in L2 (global_prefetch_b8)
#pragma unroll
    for (int j = 0; j < 4; ++j) __builtin_prefetch(bp[j] + 256, 0, 1);

    v8f c[4] = {};
#pragma unroll 2
    for (int k = 0; k < DIM; k += 32) {
        // A fragment from LDS: lanes 0-15 K=[k..k+7],[k+16..k+23]; lanes 16-31 +8
        v8bf alo = *(const v8bf*)(arow + k + half * 8);
        v8bf ahi = *(const v8bf*)(arow + k + 16 + half * 8);
        v16bf a;
#pragma unroll
        for (int i = 0; i < 8; ++i) { a[i] = alo[i]; a[i + 8] = ahi[i]; }
        // B fragments: lane holds col N, K=[k..k+7],[k+16..k+23]; lanes 16-31 +8
        const int kb = k + half * 8;
#pragma unroll
        for (int j = 0; j < 4; ++j) {
            v8bf bl = *(const v8bf*)(bp[j] + kb);
            v8bf bh = *(const v8bf*)(bp[j] + kb + 16);
            v16bf f;
#pragma unroll
            for (int i = 0; i < 8; ++i) { f[i] = bl[i]; f[i + 8] = bh[i]; }
            c[j] = __builtin_amdgcn_wmma_f32_16x16x32_bf16(
                       false, a, false, f, (short)0, c[j], false, false);
        }
    }

    // D layout: VGPR v -> row v + 8*half; col = base + (lane&15)
    const int g0 = mtile * 16;
    const int b  = g0 / SEQ;
    const int tb = g0 % SEQ;
#pragma unroll
    for (int j = 0; j < 4; ++j) {
        const int col  = nbase + j * 16 + mlane;
        const float bias = pwb[col];
#pragma unroll
        for (int v = 0; v < 8; ++v) {
            const int t = tb + v + 8 * half;
            YP[((size_t)b * LPAD + t) * DIM + col] = c[j][v] + bias;
        }
    }
}

// ---------------------------------------------------------------------------
// 5) s[b,l] = dot(YP[b,l,:], score_w)  (one 128-thread block per row)
// ---------------------------------------------------------------------------
__global__ __launch_bounds__(128) void row_score_kernel(
    const float* __restrict__ YP, const float* __restrict__ sw,
    float* __restrict__ S)
{
    const int r = blockIdx.x;                 // 0..BATCH*LPAD-1
    __shared__ float red[128];
    const float* row = YP + (size_t)r * DIM;
    float4 v = *(const float4*)(row + threadIdx.x * 4);
    float4 w = *(const float4*)(sw + threadIdx.x * 4);
    red[threadIdx.x] = v.x * w.x + v.y * w.y + v.z * w.z + v.w * w.w;
    __syncthreads();
    for (int s = 64; s > 0; s >>= 1) {
        if (threadIdx.x < s) red[threadIdx.x] += red[threadIdx.x + s];
        __syncthreads();
    }
    if (threadIdx.x == 0) S[r] = red[0];
}

// ---------------------------------------------------------------------------
// 6) scores[b,l,k] = softmax_k( blockmean_{bs=k+1}(s) + score_b )
// ---------------------------------------------------------------------------
__global__ __launch_bounds__(256) void blk_scores_kernel(
    const float* __restrict__ S, const float* __restrict__ sbp,
    float* __restrict__ SC)
{
    const int idx = blockIdx.x * 256 + threadIdx.x;
    if (idx >= BATCH * LPAD) return;
    const int b = idx / LPAD;
    const int l = idx % LPAD;
    const float* s = S + (size_t)b * LPAD;
    const float sb = sbp[0];
    float v[4];
#pragma unroll
    for (int k = 0; k < 4; ++k) {
        const int bs = k + 1;
        const int st = (l / bs) * bs;
        float m = 0.f;
        for (int j = 0; j < bs; ++j) m += s[st + j];
        v[k] = m / (float)bs + sb;
    }
    float mx = fmaxf(fmaxf(v[0], v[1]), fmaxf(v[2], v[3]));
    float e[4], z = 0.f;
#pragma unroll
    for (int k = 0; k < 4; ++k) { e[k] = __expf(v[k] - mx); z += e[k]; }
    const float inv = 1.f / z;
    float4 o = { e[0] * inv, e[1] * inv, e[2] * inv, e[3] * inv };
    *(float4*)(SC + (size_t)idx * 4) = o;
}

// ---------------------------------------------------------------------------
// 7) Row-streamed attention over scores:
//    sim(i,j)=dot4(sc_i,sc_j) in (0,1] -> exp() safe without max shift.
//    SC2[i,:] = (sum_j e_ij * sc_j) / (sum_j e_ij).  One block per (b,i).
// ---------------------------------------------------------------------------
__global__ __launch_bounds__(256) void attn_mix_kernel(
    const float* __restrict__ SC, float* __restrict__ SC2)
{
    const int b = blockIdx.x / LPAD;
    const int i = blockIdx.x % LPAD;
    const float* sc = SC + (size_t)b * LPAD * 4;
    const float4 si = *(const float4*)(sc + (size_t)i * 4);

    float z = 0.f, a0 = 0.f, a1 = 0.f, a2 = 0.f, a3 = 0.f;
    for (int j = threadIdx.x; j < LPAD; j += 256) {
        float4 sj = *(const float4*)(sc + (size_t)j * 4);
        float m = si.x * sj.x + si.y * sj.y + si.z * sj.z + si.w * sj.w;
        float e = __expf(m);
        z += e; a0 += e * sj.x; a1 += e * sj.y; a2 += e * sj.z; a3 += e * sj.w;
    }
    __shared__ float red[256];
    float vals[5] = { z, a0, a1, a2, a3 };
    float out[5];
#pragma unroll
    for (int q = 0; q < 5; ++q) {
        red[threadIdx.x] = vals[q];
        __syncthreads();
        for (int s = 128; s > 0; s >>= 1) {
            if (threadIdx.x < s) red[threadIdx.x] += red[threadIdx.x + s];
            __syncthreads();
        }
        out[q] = red[0];
        __syncthreads();
    }
    if (threadIdx.x == 0) {
        const float inv = 1.f / out[0];
        float4 o = { out[1] * inv, out[2] * inv, out[3] * inv, out[4] * inv };
        *(float4*)(SC2 + ((size_t)b * LPAD + i) * 4) = o;
    }
}

// ---------------------------------------------------------------------------
// 8) out[b,p,d] = (1/4) sum_{q<4} sum_{k<4} SC2[l,k] * blockmean_{bs=k+1}(YP)[l,d]
//    with l = 4p+q.  One 128-thread block per (b,p); float4 over d.
// ---------------------------------------------------------------------------
__global__ __launch_bounds__(128) void out_pool_kernel(
    const float* __restrict__ YP, const float* __restrict__ SC2,
    float* __restrict__ OUT)
{
    const int blk = blockIdx.x;           // b*1024 + p
    const int b = blk >> 10;
    const int p = blk & 1023;
    const int d0 = threadIdx.x * 4;
    const float* yp  = YP  + (size_t)b * LPAD * DIM;
    const float* sc2 = SC2 + (size_t)b * LPAD * 4;

    float4 acc = { 0.f, 0.f, 0.f, 0.f };
#pragma unroll
    for (int q = 0; q < 4; ++q) {
        const int l = 4 * p + q;
        const float4 wv = *(const float4*)(sc2 + (size_t)l * 4);
        const float wk[4] = { wv.x, wv.y, wv.z, wv.w };
#pragma unroll
        for (int k = 0; k < 4; ++k) {
            const int bs = k + 1;
            const int st = (l / bs) * bs;
            float4 m = { 0.f, 0.f, 0.f, 0.f };
            for (int j = 0; j < bs; ++j) {
                float4 v = *(const float4*)(yp + (size_t)(st + j) * DIM + d0);
                m.x += v.x; m.y += v.y; m.z += v.z; m.w += v.w;
            }
            const float f = wk[k] / (float)bs;
            acc.x += m.x * f; acc.y += m.y * f; acc.z += m.z * f; acc.w += m.w * f;
        }
    }
    acc.x *= 0.25f; acc.y *= 0.25f; acc.z *= 0.25f; acc.w *= 0.25f;
    *(float4*)(OUT + ((size_t)b * 1024 + p) * DIM + d0) = acc;
}

// ---------------------------------------------------------------------------
extern "C" void kernel_launch(void* const* d_in, const int* in_sizes, int n_in,
                              void* d_out, int out_size, void* d_ws, size_t ws_size,
                              hipStream_t stream)
{
    const int*   x    = (const int*)d_in[0];
    const float* emb  = (const float*)d_in[1];
    const float* dww  = (const float*)d_in[2];
    const float* dwb  = (const float*)d_in[3];
    const float* pww  = (const float*)d_in[4];
    const float* pwb  = (const float*)d_in[5];
    const float* sw   = (const float*)d_in[6];
    const float* sb   = (const float*)d_in[7];
    float* out = (float*)d_out;

    // workspace carve-up (256B aligned)
    char* ws = (char*)d_ws;
    size_t off = 0;
    auto carve = [&](size_t bytes) -> char* {
        char* p = ws + off;
        off = (off + bytes + 255) & ~(size_t)255;
        return p;
    };
    __bf16* Y   = (__bf16*)carve((size_t)MROWS * DIM * sizeof(__bf16));      // 16.8 MB
    __bf16* Wb  = (__bf16*)carve((size_t)DIM * DIM * sizeof(__bf16));        // 0.5 MB
    float*  YP  = (float*) carve((size_t)BATCH * LPAD * DIM * sizeof(float));// 33.6 MB
    float*  S   = (float*) carve((size_t)BATCH * LPAD * sizeof(float));
    float*  SC  = (float*) carve((size_t)BATCH * LPAD * 4 * sizeof(float));
    float*  SC2 = (float*) carve((size_t)BATCH * LPAD * 4 * sizeof(float));

    // 1) gather + depthwise conv -> bf16
    conv_embed_kernel<<<MROWS, 128, 0, stream>>>(x, emb, dww, dwb, Y);
    // 2) pw_w -> bf16
    cvt_pw_kernel<<<(DIM * DIM + 255) / 256, 256, 0, stream>>>(pww, Wb);
    // 3) zero pad rows of YP
    zeropad_kernel<<<(BATCH * (LPAD - SEQ) * DIM + 255) / 256, 256, 0, stream>>>(YP);
    // 4) WMMA GEMM + bias (async-LDS A staging, 16x64 per wave)
    gemm_pw_kernel<<<MROWS / 16, 256, 0, stream>>>(Y, Wb, pwb, YP);
    // 5) per-row score dot
    row_score_kernel<<<BATCH * LPAD, 128, 0, stream>>>(YP, sw, S);
    // 6) block means + k-softmax
    blk_scores_kernel<<<(BATCH * LPAD + 255) / 256, 256, 0, stream>>>(S, sb, SC);
    // 7) row-streamed attention mix
    attn_mix_kernel<<<BATCH * LPAD, 256, 0, stream>>>(SC, SC2);
    // 8) weighted block means + final 4:1 pool
    out_pool_kernel<<<BATCH * 1024, 128, 0, stream>>>(YP, SC2, out);
}
